// TinyStoriesTransformer_28965259444298
// MI455X (gfx1250) — compile-verified, matrix operand order checked by reading, not run on hardware
//
#include <hip/hip_runtime.h>
#include <cstddef>

// ============================================================================
// TinyStories transformer forward on MI455X (gfx1250, wave32, WMMA).
//
// Roofline: ~615 GFLOP of GEMM work vs ~0.5 GB of fp32 weights. All matmuls
// run on v_wmma_f32_16x16x32_bf16 (f32 accumulate). Activations are carried
// in bf16 end-to-end (every GEMM A operand / Q/K/V / ctx is only ever
// consumed as a bf16 WMMA operand), halving activation traffic and making
// A-tile staging a raw byte copy -- which lets the GEMM use the CDNA5
// GLOBAL_LOAD_ASYNC_TO_LDS path (ASYNCcnt). Weights arrive fp32 from the
// harness and are converted to bf16 while staging into LDS.
// ============================================================================

typedef __attribute__((ext_vector_type(16))) __bf16 v16bf;
typedef __attribute__((ext_vector_type(8)))  float  v8f;
typedef __attribute__((ext_vector_type(8)))  unsigned short v8u16;
typedef __attribute__((ext_vector_type(16))) unsigned short v16u16;
typedef __attribute__((ext_vector_type(4)))  int v4i;

#if __has_builtin(__builtin_amdgcn_global_load_async_to_lds_b128)
#define HAVE_ASYNC_LDS 1
// Builtin prototype (from the round-2 diagnostic): "vV4i*1V4i*3IiIi"
//   (v4i addrspace(1)* global, v4i addrspace(3)* lds, imm offset, imm cpol)
typedef __attribute__((address_space(1))) v4i* gv4ip_t;
typedef __attribute__((address_space(3))) v4i* lv4ip_t;
#else
#define HAVE_ASYNC_LDS 0
#endif

#define DEVFN static __device__ __forceinline__

DEVFN unsigned short f2bf(float f) {
  unsigned int u = __builtin_bit_cast(unsigned int, f);
  u += 0x7FFFu + ((u >> 16) & 1u);           // round-to-nearest-even
  return (unsigned short)(u >> 16);
}

DEVFN void wait_async_lds() {
#if HAVE_ASYNC_LDS
#if __has_builtin(__builtin_amdgcn_s_wait_asynccnt)
  __builtin_amdgcn_s_wait_asynccnt(0);
#else
  asm volatile("s_wait_asynccnt 0" ::: "memory");
#endif
#endif
}

// Load one 16x32 bf16 WMMA operand fragment from an LDS tile laid out
// [row][k] with row stride `ld` elements (ld*2 bytes must be a 16B multiple
// so the b128 loads stay aligned). Per CDNA5 ISA 7.12.2: lane L<16 holds row
// L with K = {0..7, 16..23}; lane L>=16 holds row L-16 with K = {8..15,
// 24..31}. A tiles are stored [m][k]; B tiles are stored [n][k] so the same
// loader serves both operands.
DEVFN v16bf load_frag(const unsigned short* lds, int ld, int row0, int k0) {
  const int lane = (int)(threadIdx.x & 31u);
  const unsigned short* p = lds + (row0 + (lane & 15)) * ld + k0 + ((lane >> 4) << 3);
  v8u16 lo = *(const v8u16*)(p);             // K = kb .. kb+7
  v8u16 hi = *(const v8u16*)(p + 16);        // K = kb+16 .. kb+23
  v16u16 full = __builtin_shufflevector(lo, hi,
      0,1,2,3,4,5,6,7,8,9,10,11,12,13,14,15);
  return __builtin_bit_cast(v16bf, full);
}

DEVFN v8f wmma_bf16(v16bf a, v16bf b, v8f c) {
  return __builtin_amdgcn_wmma_f32_16x16x32_bf16(
      /*neg_a=*/false, a, /*neg_b=*/false, b,
      /*c_mod=*/(short)0, c, /*reuse_a=*/false, /*reuse_b=*/false);
}

DEVFN v8f v8f_zero() {
  v8f z;
#pragma unroll
  for (int e = 0; e < 8; ++e) z[e] = 0.0f;
  return z;
}

// ============================================================================
// Tiled WMMA GEMM:  C[M,N] = epi( A[M,K] @ B + bias ),  A is bf16, B is fp32.
//   BT=false: B is [K,N] row-major (weights)  -> transposed into LDS [n][k]
//   BT=true : B is [N,K] row-major (tok_emb)  -> direct copy into LDS [n][k]
// EPI: 0 = +bias, bf16 out (QKV) ; 1 = +bias+residual, fp32 out (O/FF2 -> x)
//      2 = +bias + exact GELU, bf16 out (FF1) ; 3 = plain, fp32 out (LM head)
// 128x128x32 tiles, 256 threads = 8 waves (4x2), wave tile 32x64 (2x4 WMMA).
// Double-buffered LDS: A tile staged via GLOBAL_LOAD_ASYNC_TO_LDS (bf16 byte
// copy, ASYNCcnt) while B (fp32) ping-pongs through registers with bf16
// conversion; one barrier per k-step; compute overlaps the fetches.
// ============================================================================
template <int EPI, bool BT>
__global__ __launch_bounds__(256)
void gemm_wmma_k(const unsigned short* __restrict__ A, const float* __restrict__ Bmat,
                 const float* __restrict__ bias, const float* __restrict__ res,
                 void* __restrict__ Cout, int M, int N, int K) {
  constexpr bool OBF = (EPI == 0 || EPI == 2);   // bf16 output?
  __shared__ __align__(16) unsigned short lsA[2][128 * 40];
  __shared__ __align__(16) unsigned short lsB[2][128 * 40];

  const int tid  = (int)threadIdx.x;
  const int n0   = (int)blockIdx.x * 128;
  const int m0   = (int)blockIdx.y * 128;
  const int wave = tid >> 5;
  const int wm   = (wave >> 1) * 32;
  const int wn   = (wave & 1) * 64;
  const int arow = tid >> 1;           // 0..127
  const int acol = (tid & 1) * 16;     // 0 or 16
  const int brow = tid >> 3;           // 0..31  (k in tile, BT=false)
  const int bcol = (tid & 7) * 16;     // n in tile (BT=false)

  v8f acc[2][4];
#pragma unroll
  for (int i = 0; i < 2; ++i)
#pragma unroll
    for (int j = 0; j < 4; ++j) acc[i][j] = v8f_zero();

  float4 breg[4];

  // A tile: raw bf16 copy, 32B per thread.
  auto issueA = [&](int t, int buf) {
    const unsigned short* src = A + (size_t)(m0 + arow) * K + t * 32 + acol;
    unsigned short* dst = &lsA[buf][arow * 40 + acol];
#if HAVE_ASYNC_LDS
    __builtin_amdgcn_global_load_async_to_lds_b128((gv4ip_t)src, (lv4ip_t)dst, 0, 0);
    __builtin_amdgcn_global_load_async_to_lds_b128((gv4ip_t)(src + 8), (lv4ip_t)(dst + 8), 0, 0);
#else
    v8u16 x0 = *(const v8u16*)src;
    v8u16 x1 = *(const v8u16*)(src + 8);
    *(v8u16*)dst = x0;
    *(v8u16*)(dst + 8) = x1;
#endif
  };
  // B tile: fp32 loads into registers (latency hidden under the WMMAs).
  auto issueB = [&](int t) {
    const float4* bp = BT
        ? (const float4*)(Bmat + (size_t)(n0 + arow) * K + t * 32 + acol)
        : (const float4*)(Bmat + (size_t)(t * 32 + brow) * N + n0 + bcol);
#pragma unroll
    for (int u = 0; u < 4; ++u) breg[u] = bp[u];
  };
  // B commit: fp32 -> bf16 into LDS [n][k].
  auto commitB = [&](int buf) {
    if (BT) {
#pragma unroll
      for (int u = 0; u < 4; ++u) {
        unsigned short* d = &lsB[buf][arow * 40 + acol + u * 4];
        d[0] = f2bf(breg[u].x); d[1] = f2bf(breg[u].y);
        d[2] = f2bf(breg[u].z); d[3] = f2bf(breg[u].w);
      }
    } else {
#pragma unroll
      for (int u = 0; u < 4; ++u) {
        const int n = bcol + u * 4;
        lsB[buf][(n + 0) * 40 + brow] = f2bf(breg[u].x);
        lsB[buf][(n + 1) * 40 + brow] = f2bf(breg[u].y);
        lsB[buf][(n + 2) * 40 + brow] = f2bf(breg[u].z);
        lsB[buf][(n + 3) * 40 + brow] = f2bf(breg[u].w);
      }
    }
  };

  const int nT = K >> 5;
  issueB(0);
  issueA(0, 0);
  commitB(0);
  wait_async_lds();
  __syncthreads();

  for (int t = 0; t < nT; ++t) {
    const int cur = t & 1, nxt = cur ^ 1;
    const bool more = (t + 1 < nT);
    if (more) {
      issueA(t + 1, nxt);   // async copy straight into the other LDS buffer
      issueB(t + 1);        // fp32 weights into registers
      __builtin_prefetch(BT ? (const void*)(Bmat + (size_t)(n0 + arow) * K + (t + 2) * 32)
                            : (const void*)(Bmat + (size_t)((t + 2) * 32 + brow) * N + n0 + bcol),
                         0, 0);
    }
    v16bf a0 = load_frag(lsA[cur], 40, wm, 0);
    v16bf a1 = load_frag(lsA[cur], 40, wm + 16, 0);
#pragma unroll
    for (int j = 0; j < 4; ++j) {
      v16bf b = load_frag(lsB[cur], 40, wn + j * 16, 0);
      acc[0][j] = wmma_bf16(a0, b, acc[0][j]);
      acc[1][j] = wmma_bf16(a1, b, acc[1][j]);
    }
    if (more) commitB(nxt);
    wait_async_lds();
    __syncthreads();
  }

  // epilogue: C/D layout (ISA 7.12.2): lane holds column n = lane&15,
  // VGPR r holds row r (lanes 0-15) / r+8 (lanes 16-31)
  const int lane = tid & 31;
  const int ln15 = lane & 15;
  const int hib  = lane >> 4;
  float* Cf = (float*)Cout;
  unsigned short* Cb = (unsigned short*)Cout;
#pragma unroll
  for (int i = 0; i < 2; ++i) {
#pragma unroll
    for (int j = 0; j < 4; ++j) {
      const int n = n0 + wn + j * 16 + ln15;
      const float bv = (EPI == 3) ? 0.0f : bias[n];
#pragma unroll
      for (int r = 0; r < 8; ++r) {
        const int m = m0 + wm + i * 16 + hib * 8 + r;
        float v = acc[i][j][r] + bv;
        if (EPI == 1) v += res[(size_t)m * N + n];
        if (EPI == 2) v = 0.5f * v * (1.0f + erff(v * 0.70710678118654752f));
        if (OBF) Cb[(size_t)m * N + n] = f2bf(v);
        else     Cf[(size_t)m * N + n] = v;
      }
    }
  }
}

// ============================================================================
// Flash-style attention, one block per (batch, head, 64-row q-block).
// 128 threads = 4 waves; wave w owns q rows [w*16, w*16+16).
// Q/K/V are bf16 [B*S, H*dk] (dk = 64); ctx is written bf16.
// K staged [key][d] (natural B-fragment layout for Q.K^T); V staged
// transposed [d][key] for P.V. Masked entries get the reference's finite
// -1e4 fill, handled naturally by the online softmax.
// ============================================================================
__global__ __launch_bounds__(128)
void attn_wmma_k(const unsigned short* __restrict__ Q, const unsigned short* __restrict__ Kb,
                 const unsigned short* __restrict__ Vb, unsigned short* __restrict__ O,
                 int S, int Hn, int isLocal, int win) {
  const int dk = 64;
  const int D  = Hn * dk;
  const int qb = (int)blockIdx.x;
  const int h  = (int)blockIdx.y;
  const int b  = (int)blockIdx.z;
  const int q0 = qb * 64;
  const size_t rowBase = (size_t)b * S;
  const int colBase = h * dk;
  const float scale = 0.125f;          // 1/sqrt(64)

  __shared__ __align__(16) unsigned short lsQ[64 * 72];   // [q][d]
  __shared__ __align__(16) unsigned short lsK[64 * 72];   // [key][d]
  __shared__ __align__(16) unsigned short lsVt[64 * 72];  // [d][key]
  __shared__ __align__(16) unsigned short lsP[64 * 72];   // [q][key]

  const int tid  = (int)threadIdx.x;
  const int wave = tid >> 5;
  const int lane = tid & 31;
  const int ln15 = lane & 15;
  const int hib  = lane >> 4;
  const int r    = tid >> 1;           // staging row 0..63
  const int c    = (tid & 1) * 32;     // staging col 0 / 32

  // stage Q once (raw bf16 copy)
  {
    const unsigned short* qs = Q + (rowBase + q0 + r) * D + colBase + c;
#pragma unroll
    for (int u = 0; u < 4; ++u)
      *(v8u16*)&lsQ[r * 72 + c + u * 8] = *(const v8u16*)(qs + u * 8);
  }

  float mstat[8], lstat[8];
  v8f accO[4];
#pragma unroll
  for (int q = 0; q < 8; ++q) { mstat[q] = -1e30f; lstat[q] = 0.0f; }
#pragma unroll
  for (int j = 0; j < 4; ++j) accO[j] = v8f_zero();

  int kbBeg = 0;
  if (isLocal) { int lo = q0 - win; if (lo < 0) lo = 0; kbBeg = lo >> 6; }
  const int kbEnd = qb + 1;            // causal: key blocks <= q block
  __syncthreads();

  for (int kblk = kbBeg; kblk < kbEnd; ++kblk) {
    const int k0 = kblk * 64;
    // stage K [key][d] (copy) and V transposed [d][key]
    {
      const unsigned short* ks = Kb + (rowBase + k0 + r) * D + colBase + c;
      const unsigned short* vs = Vb + (rowBase + k0 + r) * D + colBase + c;
#pragma unroll
      for (int u = 0; u < 4; ++u) {
        *(v8u16*)&lsK[r * 72 + c + u * 8] = *(const v8u16*)(ks + u * 8);
        v8u16 vv = *(const v8u16*)(vs + u * 8);
#pragma unroll
        for (int e = 0; e < 8; ++e) lsVt[(c + u * 8 + e) * 72 + r] = vv[e];
      }
    }
    __syncthreads();

    // S = Q K^T (16 q rows x 64 keys per wave)
    v16bf qa0 = load_frag(lsQ, 72, wave * 16, 0);
    v16bf qa1 = load_frag(lsQ, 72, wave * 16, 32);
    float pv[4][8];
    float rowmax[8];
#pragma unroll
    for (int q = 0; q < 8; ++q) rowmax[q] = -1e30f;
#pragma unroll
    for (int j = 0; j < 4; ++j) {
      v16bf kf0 = load_frag(lsK, 72, j * 16, 0);
      v16bf kf1 = load_frag(lsK, 72, j * 16, 32);
      v8f sc = v8f_zero();
      sc = wmma_bf16(qa0, kf0, sc);
      sc = wmma_bf16(qa1, kf1, sc);
      const int kkey = k0 + j * 16 + ln15;
#pragma unroll
      for (int q = 0; q < 8; ++q) {
        const int qq = q0 + wave * 16 + hib * 8 + q;
        float s = sc[q] * scale;
        const bool ok = (kkey <= qq) && (!isLocal || kkey >= qq - win);
        s = ok ? s : -10000.0f;
        pv[j][q] = s;
        rowmax[q] = fmaxf(rowmax[q], s);
      }
    }
    // half-wave row reductions (lanes 0-15 share rows 0-7; 16-31 rows 8-15)
#pragma unroll
    for (int q = 0; q < 8; ++q) {
      float m = rowmax[q];
      m = fmaxf(m, __shfl_xor(m, 8, 16));
      m = fmaxf(m, __shfl_xor(m, 4, 16));
      m = fmaxf(m, __shfl_xor(m, 2, 16));
      m = fmaxf(m, __shfl_xor(m, 1, 16));
      rowmax[q] = m;
    }
    float alpha[8], rsum[8];
#pragma unroll
    for (int q = 0; q < 8; ++q) {
      const float mn = fmaxf(mstat[q], rowmax[q]);
      alpha[q] = __expf(mstat[q] - mn);
      mstat[q] = mn;
      rsum[q] = 0.0f;
    }
#pragma unroll
    for (int j = 0; j < 4; ++j)
#pragma unroll
      for (int q = 0; q < 8; ++q) {
        const float p = __expf(pv[j][q] - mstat[q]);
        pv[j][q] = p;
        rsum[q] += p;
      }
#pragma unroll
    for (int q = 0; q < 8; ++q) {
      float s = rsum[q];
      s += __shfl_xor(s, 8, 16);
      s += __shfl_xor(s, 4, 16);
      s += __shfl_xor(s, 2, 16);
      s += __shfl_xor(s, 1, 16);
      lstat[q] = lstat[q] * alpha[q] + s;
    }
#pragma unroll
    for (int j = 0; j < 4; ++j)
#pragma unroll
      for (int q = 0; q < 8; ++q) accO[j][q] *= alpha[q];

    // reshape P (C-layout) -> A-layout via LDS, then O += P V
#pragma unroll
    for (int j = 0; j < 4; ++j)
#pragma unroll
      for (int q = 0; q < 8; ++q)
        lsP[(wave * 16 + hib * 8 + q) * 72 + j * 16 + ln15] = f2bf(pv[j][q]);
    __syncthreads();

    v16bf pa0 = load_frag(lsP, 72, wave * 16, 0);
    v16bf pa1 = load_frag(lsP, 72, wave * 16, 32);
#pragma unroll
    for (int j = 0; j < 4; ++j) {
      v16bf vf0 = load_frag(lsVt, 72, j * 16, 0);
      v16bf vf1 = load_frag(lsVt, 72, j * 16, 32);
      accO[j] = wmma_bf16(pa0, vf0, accO[j]);
      accO[j] = wmma_bf16(pa1, vf1, accO[j]);
    }
    __syncthreads();  // before K/V/P tiles are overwritten
  }

  // ctx = O / l  (bf16)
#pragma unroll
  for (int j = 0; j < 4; ++j) {
    const int d = colBase + j * 16 + ln15;
#pragma unroll
    for (int q = 0; q < 8; ++q) {
      const int m = q0 + wave * 16 + hib * 8 + q;
      O[(rowBase + m) * D + d] = f2bf(accO[j][q] / lstat[q]);
    }
  }
}

// ============================================================================
// LayerNorm: one 256-thread block per row (D = 1024, 4 floats/thread);
// writes bf16 (consumed only as WMMA A operand).
// ============================================================================
__global__ __launch_bounds__(256)
void ln_k(const float* __restrict__ x, const float* __restrict__ w,
          const float* __restrict__ bvec, unsigned short* __restrict__ y, int D) {
  const int row = (int)blockIdx.x;
  const int tid = (int)threadIdx.x;
  const int lane = tid & 31, wave = tid >> 5;
  const float4 v = ((const float4*)(x + (size_t)row * D))[tid];

  __shared__ float red[8];
  float s = v.x + v.y + v.z + v.w;
  s += __shfl_xor(s, 16, 32); s += __shfl_xor(s, 8, 32);
  s += __shfl_xor(s, 4, 32);  s += __shfl_xor(s, 2, 32);
  s += __shfl_xor(s, 1, 32);
  if (lane == 0) red[wave] = s;
  __syncthreads();
  float tot = 0.0f;
#pragma unroll
  for (int i = 0; i < 8; ++i) tot += red[i];
  const float mean = tot / (float)D;
  __syncthreads();

  const float d0 = v.x - mean, d1 = v.y - mean, d2 = v.z - mean, d3 = v.w - mean;
  float ss = d0 * d0 + d1 * d1 + d2 * d2 + d3 * d3;
  ss += __shfl_xor(ss, 16, 32); ss += __shfl_xor(ss, 8, 32);
  ss += __shfl_xor(ss, 4, 32);  ss += __shfl_xor(ss, 2, 32);
  ss += __shfl_xor(ss, 1, 32);
  if (lane == 0) red[wave] = ss;
  __syncthreads();
  float tss = 0.0f;
#pragma unroll
  for (int i = 0; i < 8; ++i) tss += red[i];
  const float inv = rsqrtf(tss / (float)D + 1e-5f);

  const int c = tid * 4;
  unsigned short* o = y + (size_t)row * D + c;
  o[0] = f2bf(d0 * inv * w[c + 0] + bvec[c + 0]);
  o[1] = f2bf(d1 * inv * w[c + 1] + bvec[c + 1]);
  o[2] = f2bf(d2 * inv * w[c + 2] + bvec[c + 2]);
  o[3] = f2bf(d3 * inv * w[c + 3] + bvec[c + 3]);
}

// ============================================================================
// Embedding: x[row] = tok_emb[ids[row]] + pos_emb[row % S]  (fp32 residual)
// ============================================================================
__global__ __launch_bounds__(256)
void embed_k(const int* __restrict__ ids, const float* __restrict__ tok,
             const float* __restrict__ pos, float* __restrict__ x,
             int S, int D) {
  const int row = (int)blockIdx.x;
  const int s = row % S;
  const int id = ids[row];
  const float4 t = ((const float4*)(tok + (size_t)id * D))[threadIdx.x];
  const float4 p = ((const float4*)(pos + (size_t)s * D))[threadIdx.x];
  float4 o;
  o.x = t.x + p.x; o.y = t.y + p.y; o.z = t.z + p.z; o.w = t.w + p.w;
  ((float4*)(x + (size_t)row * D))[threadIdx.x] = o;
}

// ============================================================================
// Orchestration
// ============================================================================
extern "C" void kernel_launch(void* const* d_in, const int* in_sizes, int n_in,
                              void* d_out, int out_size, void* d_ws, size_t ws_size,
                              hipStream_t stream) {
  (void)in_sizes; (void)n_in; (void)out_size; (void)ws_size;
  constexpr int Bn = 2, Sn = 1024, Dn = 1024, Hn = 16, Ln = 8, FFn = 4096,
                Vn = 32000, WINn = 128;
  constexpr int Mn = Bn * Sn;  // 2048 token rows

  const int*   ids  = (const int*)  d_in[0];
  const float* tok  = (const float*)d_in[1];
  const float* pos  = (const float*)d_in[2];
  const float* Wq   = (const float*)d_in[3];
  const float* bq   = (const float*)d_in[4];
  const float* Wk   = (const float*)d_in[5];
  const float* bk   = (const float*)d_in[6];
  const float* Wv   = (const float*)d_in[7];
  const float* bv   = (const float*)d_in[8];
  const float* Wo   = (const float*)d_in[9];
  const float* bo   = (const float*)d_in[10];
  const float* ln1w = (const float*)d_in[11];
  const float* ln1b = (const float*)d_in[12];
  const float* ln2w = (const float*)d_in[13];
  const float* ln2b = (const float*)d_in[14];
  const float* W1   = (const float*)d_in[15];
  const float* b1   = (const float*)d_in[16];
  const float* W2   = (const float*)d_in[17];
  const float* b2   = (const float*)d_in[18];
  const float* lnfw = (const float*)d_in[19];
  const float* lnfb = (const float*)d_in[20];
  float* out = (float*)d_out;

  // workspace layout: x fp32 (8MB); h/q/k/v/ctx bf16 (4MB each); ff bf16 (16MB)
  const size_t MD = (size_t)Mn * Dn;
  char* base = (char*)d_ws;
  float*          x   = (float*)base;                                  // 4*MD B
  unsigned short* hb  = (unsigned short*)(base + 4 * MD);              // 2*MD B
  unsigned short* qb_ = (unsigned short*)(base + 6 * MD);
  unsigned short* kb_ = (unsigned short*)(base + 8 * MD);
  unsigned short* vb_ = (unsigned short*)(base + 10 * MD);
  unsigned short* ctx = (unsigned short*)(base + 12 * MD);
  unsigned short* ffb = (unsigned short*)(base + 14 * MD);             // 2*M*FF B

  embed_k<<<Mn, 256, 0, stream>>>(ids, tok, pos, x, Sn, Dn);

  const dim3 gD(Dn / 128, Mn / 128);    // (8, 16)
  const dim3 gFF(FFn / 128, Mn / 128);  // (32, 16)
  const dim3 gA(Sn / 64, Hn, Bn);       // (16, 16, 2)

  for (int l = 0; l < Ln; ++l) {
    const size_t wOff = (size_t)l * Dn * Dn;
    const size_t bOff = (size_t)l * Dn;
    ln_k<<<Mn, 256, 0, stream>>>(x, ln1w + bOff, ln1b + bOff, hb, Dn);
    gemm_wmma_k<0, false><<<gD, 256, 0, stream>>>(hb, Wq + wOff, bq + bOff, nullptr, qb_, Mn, Dn, Dn);
    gemm_wmma_k<0, false><<<gD, 256, 0, stream>>>(hb, Wk + wOff, bk + bOff, nullptr, kb_, Mn, Dn, Dn);
    gemm_wmma_k<0, false><<<gD, 256, 0, stream>>>(hb, Wv + wOff, bv + bOff, nullptr, vb_, Mn, Dn, Dn);
    attn_wmma_k<<<gA, 128, 0, stream>>>(qb_, kb_, vb_, ctx, Sn, Hn, l & 1, WINn);
    gemm_wmma_k<1, false><<<gD, 256, 0, stream>>>(ctx, Wo + wOff, bo + bOff, x, x, Mn, Dn, Dn);
    ln_k<<<Mn, 256, 0, stream>>>(x, ln2w + bOff, ln2b + bOff, hb, Dn);
    gemm_wmma_k<2, false><<<gFF, 256, 0, stream>>>(hb, W1 + (size_t)l * Dn * FFn, b1 + (size_t)l * FFn,
                                                   nullptr, ffb, Mn, FFn, Dn);
    gemm_wmma_k<1, false><<<gD, 256, 0, stream>>>(ffb, W2 + (size_t)l * FFn * Dn, b2 + bOff, x, x,
                                                  Mn, Dn, FFn);
  }

  ln_k<<<Mn, 256, 0, stream>>>(x, lnfw, lnfb, hb, Dn);
  // tied LM head: logits = h @ tok_emb^T  (tok_emb is [V, D] = [N][K])
  gemm_wmma_k<3, true><<<dim3(Vn / 128, Mn / 128), 256, 0, stream>>>(
      hb, tok, nullptr, nullptr, (void*)out, Mn, Vn, Dn);
}
